// modi_cgcnn_edge_46248207843555
// MI455X (gfx1250) — matrix operand order repaired and checked by reading, not compiled, not run on hardware
//
#include <hip/hip_runtime.h>
#include <hip/hip_fp16.h>

typedef __attribute__((ext_vector_type(16))) _Float16 v16h;
typedef __attribute__((ext_vector_type(8)))  float    v8f;

#define NUM_CRYSTALS 2048
#define EPSF 1e-5f
#define INV_SQRT_2F 0.70710678118654752440f

union V16H { v16h v; _Float16 h[16]; };
union V8F  { v8f  v; float    f[8];  };
union H8   { uint4 u; _Float16 h[8]; };

__device__ __forceinline__ float sigmoidf_(float x) { return 1.0f / (1.0f + __expf(-x)); }
__device__ __forceinline__ float siluf_(float x)    { return x * sigmoidf_(x); }

// ---------------------------------------------------------------------------
// Zero a float region of the workspace (stat accumulators).
__global__ __launch_bounds__(256) void k_zero(float* __restrict__ p, long n) {
    long i = (long)blockIdx.x * 256 + threadIdx.x;
    if (i < n) p[i] = 0.0f;
}

// Edge counts per crystal; LDS pre-reduction on the block's base crystal
// (seg is sorted, so nearly all 256 edges of a block share one crystal).
__global__ __launch_bounds__(256) void k_count(const int* __restrict__ seg,
                                               float* __restrict__ cnt, int E) {
    __shared__ float cL;
    __shared__ int   sb;
    int tid = threadIdx.x;
    long e0 = (long)blockIdx.x * 256;
    if (tid == 0) { cL = 0.0f; sb = seg[(e0 < E) ? e0 : (long)(E - 1)]; }
    __syncthreads();
    long e = e0 + tid;
    if (e < E) {
        int s = seg[e];
        if (s == sb) atomicAdd(&cL, 1.0f);
        else         atomicAdd(&cnt[s], 1.0f);
    }
    __syncthreads();
    if (tid == 0) atomicAdd(&cnt[sb], cL);
}

// mean / inv-std finalization for segment stats.
__global__ __launch_bounds__(256) void k_finalize(const float* __restrict__ cnt,
                                                  const float* __restrict__ sum,
                                                  const float* __restrict__ sq,
                                                  float* __restrict__ mean,
                                                  float* __restrict__ istd, int feat) {
    long i = (long)blockIdx.x * 256 + threadIdx.x;
    if (i < (long)NUM_CRYSTALS * feat) {
        float c  = fmaxf(cnt[i / feat], 1.0f);
        float mu = sum[i] / c;
        float va = sq[i] / c - mu * mu;
        mean[i] = mu;
        istd[i] = rsqrtf(fmaxf(va, 0.0f) + EPSF);
    }
}

// ---------------------------------------------------------------------------
// Kernel 1: fused endpoint gather + [E,192] x [192,128] GEMM via WMMA f16.
// Block = 256 threads = 8 waves; each wave owns a 16-edge M-tile (128 edges/block).
// W_full lives in LDS as f16, N-major, so B fragments are 2x ds_load_b128.
// crystal_norm1 stats are pre-reduced in LDS (ds_add_f32) on the block's base
// crystal; boundary edges leak to global atomics.
__global__ __launch_bounds__(256)
void k1_gemm_stats(const float* __restrict__ atom, const float* __restrict__ nbr,
                   const int* __restrict__ idx, const int* __restrict__ seg,
                   const float* __restrict__ Wfull,
                   _Float16* __restrict__ gated,
                   float* __restrict__ sum1, float* __restrict__ sq1, int E) {
    __shared__ _Float16 Wt[128 * 192];  // Wt[n][k], n-major for contiguous-K B reads
    __shared__ float    sumL[128], sqL[128];
    __shared__ int      sBase;
    int tid = threadIdx.x;
    long e0 = (long)blockIdx.x * 128;
    for (int i = tid; i < 128 * 192; i += 256) {
        int n = i / 192, k = i % 192;
        Wt[i] = (_Float16)Wfull[k * 128 + n];
    }
    if (tid < 128) { sumL[tid] = 0.0f; sqL[tid] = 0.0f; }
    if (tid == 0)  sBase = seg[(e0 < E) ? e0 : (long)(E - 1)];
    __syncthreads();
    int sb = sBase;

    int  w = tid >> 5, l = tid & 31;
    int  m = l & 15, half = l >> 4;
    long e  = e0 + w * 16 + m;
    long ec = (e < E) ? e : (long)(E - 1);
    int  i0 = idx[ec * 2 + 0];
    int  i1 = idx[ec * 2 + 1];

    V8F acc[8];
#pragma unroll
    for (int nt = 0; nt < 8; ++nt)
#pragma unroll
        for (int r = 0; r < 8; ++r) acc[nt].f[r] = 0.0f;

#pragma unroll
    for (int ks = 0; ks < 6; ++ks) {
        int  kb = ks * 32;
        V16H a;
        // A fragment, contiguous K order: half 0 holds K {kb..kb+15},
        // half 1 holds K {kb+16..kb+31}; 8-float segments never straddle
        // the 64-wide concat regions, so each is two aligned float4 loads.
#pragma unroll
        for (int s8 = 0; s8 < 2; ++s8) {
            int kk = kb + half * 16 + s8 * 8;
            const float* src;
            if (kk < 64)       src = atom + (long)i0 * 64 + kk;
            else if (kk < 128) src = atom + (long)i1 * 64 + (kk - 64);
            else               src = nbr + ec * 64 + (kk - 128);
            float4 p0 = *(const float4*)src;
            float4 p1 = *(const float4*)(src + 4);
            int o = s8 * 8;
            a.h[o + 0] = (_Float16)p0.x; a.h[o + 1] = (_Float16)p0.y;
            a.h[o + 2] = (_Float16)p0.z; a.h[o + 3] = (_Float16)p0.w;
            a.h[o + 4] = (_Float16)p1.x; a.h[o + 5] = (_Float16)p1.y;
            a.h[o + 6] = (_Float16)p1.z; a.h[o + 7] = (_Float16)p1.w;
        }
#pragma unroll
        for (int nt = 0; nt < 8; ++nt) {
            int n = nt * 16 + m;
            const v16h b = *(const v16h*)&Wt[n * 192 + kb + half * 16];
            acc[nt].v = __builtin_amdgcn_wmma_f32_16x16x32_f16(
                false, a.v, false, b, (short)0, acc[nt].v, false, false);
        }
    }

    // D layout: lane half selects M = r or r+8; N = nt*16 + (lane&15).
    int sv[8];
#pragma unroll
    for (int r = 0; r < 8; ++r) {
        long er = e0 + w * 16 + r + 8 * half;
        sv[r] = seg[(er < E) ? er : (long)(E - 1)];
    }
#pragma unroll
    for (int nt = 0; nt < 8; ++nt) {
        int f = nt * 16 + m;
#pragma unroll
        for (int r = 0; r < 8; ++r) {
            long er = e0 + w * 16 + r + 8 * half;
            if (er < E) {
                float v = acc[nt].f[r];
                gated[er * 128 + f] = (_Float16)v;
                if (sv[r] == sb) {
                    atomicAdd(&sumL[f], v);
                    atomicAdd(&sqL[f], v * v);
                } else {
                    atomicAdd(&sum1[(long)sv[r] * 128 + f], v);
                    atomicAdd(&sq1[(long)sv[r] * 128 + f], v * v);
                }
            }
        }
    }
    __syncthreads();
    if (tid < 128) {
        atomicAdd(&sum1[(long)sb * 128 + tid], sumL[tid]);
        atomicAdd(&sq1[(long)sb * 128 + tid], sqL[tid]);
    }
}

// ---------------------------------------------------------------------------
// Kernel 2: crystal_norm1 + sigmoid gate + row layernorm + SiLU + crystal_norm2 stats.
// One edge per thread; gated read as f16. Stats pre-reduced in LDS on the base crystal.
__global__ __launch_bounds__(256)
void k2_gate(const _Float16* __restrict__ gated, const int* __restrict__ seg,
             const float* __restrict__ mean1, const float* __restrict__ istd1,
             const float* __restrict__ n1w, const float* __restrict__ n1b,
             const float* __restrict__ Wm, const float* __restrict__ lnw,
             const float* __restrict__ lnb,
             float* __restrict__ sum2, float* __restrict__ sq2,
             _Float16* __restrict__ xout, int E) {
    __shared__ float s_w[128], s_b[128], s_wm[64], s_lw[64], s_lb[64];
    __shared__ float sum2L[64], sq2L[64];
    __shared__ int   sBase;
    int tid = threadIdx.x;
    long e0 = (long)blockIdx.x * 256;
    if (tid < 128) { s_w[tid] = n1w[tid]; s_b[tid] = n1b[tid]; }
    if (tid < 64)  { s_wm[tid] = Wm[tid]; s_lw[tid] = lnw[tid]; s_lb[tid] = lnb[tid];
                     sum2L[tid] = 0.0f; sq2L[tid] = 0.0f; }
    if (tid == 0)  sBase = seg[(e0 < E) ? e0 : (long)(E - 1)];
    __syncthreads();
    int sb = sBase;

    long e   = e0 + tid;
    bool act = (e < E);
    long ec  = act ? e : (long)(E - 1);
    if (e + 256 < E) __builtin_prefetch(gated + (e + 256) * 128, 0, 0);

    int s = seg[ec];
    const _Float16* g = gated + ec * 128;
    const float* mr = mean1 + (long)s * 128;
    const float* ir = istd1 + (long)s * 128;

    float dot = 0.0f;
    for (int f = 64; f < 128; ++f) {
        float gv = ((float)g[f] - mr[f]) * ir[f] * s_w[f] + s_b[f];
        dot += gv * s_wm[f - 64];
    }
    float filt = sigmoidf_(dot);

    float core[64], ms = 0.0f, vs = 0.0f;
    for (int f = 0; f < 64; ++f) {
        float gv = ((float)g[f] - mr[f]) * ir[f] * s_w[f] + s_b[f];
        core[f] = gv; ms += gv; vs += gv * gv;
    }
    float mu = ms * (1.0f / 64.0f);
    float va = vs * (1.0f / 64.0f) - mu * mu;
    float is = rsqrtf(fmaxf(va, 0.0f) + EPSF);

    _Float16* xr = xout + ec * 64;
    for (int f = 0; f < 64; ++f) {
        float cn = (core[f] - mu) * is * s_lw[f] + s_lb[f];
        float xv = filt * siluf_(cn);
        if (act) {
            xr[f] = (_Float16)xv;
            if (s == sb) {
                atomicAdd(&sum2L[f], xv);
                atomicAdd(&sq2L[f], xv * xv);
            } else {
                atomicAdd(&sum2[(long)s * 64 + f], xv);
                atomicAdd(&sq2[(long)s * 64 + f], xv * xv);
            }
        }
    }
    __syncthreads();
    if (tid < 64) {
        atomicAdd(&sum2[(long)sb * 64 + tid], sum2L[tid]);
        atomicAdd(&sq2[(long)sb * 64 + tid], sq2L[tid]);
    }
}

// ---------------------------------------------------------------------------
// Kernel 3: crystal_norm2 + two WMMA residual blocks (64->32->64) + final blend.
// Raw x rows are staged into LDS with gfx1250 async load-to-LDS DMA (ASYNCcnt),
// then normalized into the per-wave 16x64 f16 WMMA tile.
__global__ __launch_bounds__(256)
void k3_residual(const _Float16* __restrict__ xbuf, const int* __restrict__ seg,
                 const float* __restrict__ mean2, const float* __restrict__ istd2,
                 const float* __restrict__ n2w, const float* __restrict__ n2b,
                 const float* __restrict__ W1a, const float* __restrict__ b1a,
                 const float* __restrict__ W2a, const float* __restrict__ b2a,
                 const float* __restrict__ W1b, const float* __restrict__ b1b,
                 const float* __restrict__ W2b, const float* __restrict__ b2b,
                 const float* __restrict__ nbr, float* __restrict__ out, int E) {
    __shared__ _Float16 w1t[2][32 * 64];   // [n][k]
    __shared__ _Float16 w2t[2][64 * 32];   // [n][k]
    __shared__ float    b1L[2][32], b2L[2][64], n2wL[64], n2bL[64];
    __shared__ _Float16 xs[8][16][64];     // async-staged raw x rows
    __shared__ _Float16 xt[8][16][64];     // normalized WMMA tile
    __shared__ _Float16 ht[8][16][32];
    int tid = threadIdx.x;
    for (int i = tid; i < 2048; i += 256) {
        int n1 = i >> 6, k1 = i & 63;       // 32x64
        w1t[0][i] = (_Float16)W1a[k1 * 32 + n1];
        w1t[1][i] = (_Float16)W1b[k1 * 32 + n1];
        int n2 = i >> 5, k2 = i & 31;       // 64x32
        w2t[0][i] = (_Float16)W2a[k2 * 64 + n2];
        w2t[1][i] = (_Float16)W2b[k2 * 64 + n2];
    }
    if (tid < 32) { b1L[0][tid] = b1a[tid]; b1L[1][tid] = b1b[tid]; }
    if (tid < 64) { b2L[0][tid] = b2a[tid]; b2L[1][tid] = b2b[tid];
                    n2wL[tid] = n2w[tid];   n2bL[tid] = n2b[tid]; }

    int  w = tid >> 5, l = tid & 31, m = l & 15, half = l >> 4;
    long e0 = (long)blockIdx.x * 128;
    long e  = e0 + w * 16 + m;
    long ec = (e < E) ? e : (long)(E - 1);
    int  s  = seg[ec];

    // Async-DMA the wave's 16x64 f16 tile (2 KB) from global into LDS:
    // 4 x global_load_async_to_lds_b128 per lane, tracked by ASYNCcnt.
    {
        unsigned lds_base = (unsigned)(uintptr_t)(&xs[w][0][0]);
#pragma unroll
        for (int j = 0; j < 4; ++j) {
            int  t    = j * 512 + l * 16;   // byte offset within the 2 KB tile
            int  row  = t >> 7;             // 128 B per edge row
            int  colb = t & 127;
            long er   = e0 + w * 16 + row;
            if (er >= E) er = (long)(E - 1);
            unsigned long long gsrc =
                (unsigned long long)(uintptr_t)((const char*)xbuf + er * 128 + colb);
            unsigned ldst = lds_base + (unsigned)(row * 128 + colb);
            asm volatile("global_load_async_to_lds_b128 %0, %1, off"
                         :: "v"(ldst), "v"(gsrc) : "memory");
        }
        asm volatile("s_wait_asynccnt 0x0" ::: "memory");
    }
    __syncthreads();

    // Normalize (crystal_norm2) out of the staged LDS copy into the WMMA tile.
    {
        const _Float16* xr = &xs[w][m][half * 32];
        const float* m2 = mean2 + (long)s * 64 + half * 32;
        const float* i2 = istd2 + (long)s * 64 + half * 32;
#pragma unroll
        for (int j0 = 0; j0 < 32; j0 += 8) {
            H8 xh; xh.u = *(const uint4*)(xr + j0);
#pragma unroll
            for (int j = 0; j < 8; ++j) {
                int f = half * 32 + j0 + j;
                float xn = ((float)xh.h[j] - m2[j0 + j]) * i2[j0 + j] * n2wL[f] + n2bL[f];
                xt[w][m][f] = (_Float16)xn;
            }
        }
    }
    __syncthreads();

#pragma unroll
    for (int rb = 0; rb < 2; ++rb) {
        // h = silu(x @ W1 + b1)   (K=64 in two 32-chunks, N=32 in two tiles)
        V8F accH[2];
#pragma unroll
        for (int nt = 0; nt < 2; ++nt) {
            float bv = b1L[rb][nt * 16 + m];
#pragma unroll
            for (int r = 0; r < 8; ++r) accH[nt].f[r] = bv;
        }
#pragma unroll
        for (int c = 0; c < 2; ++c) {
            int  kb = c * 32;
            V16H a;
#pragma unroll
            for (int j = 0; j < 8; ++j) {
                a.h[j]     = xt[w][m][kb + half * 16 + j];
                a.h[8 + j] = xt[w][m][kb + half * 16 + 8 + j];
            }
#pragma unroll
            for (int nt = 0; nt < 2; ++nt) {
                int n = nt * 16 + m;
                const v16h b = *(const v16h*)&w1t[rb][n * 64 + kb + half * 16];
                accH[nt].v = __builtin_amdgcn_wmma_f32_16x16x32_f16(
                    false, a.v, false, b, (short)0, accH[nt].v, false, false);
            }
        }
#pragma unroll
        for (int nt = 0; nt < 2; ++nt)
#pragma unroll
            for (int r = 0; r < 8; ++r)
                ht[w][r + 8 * half][nt * 16 + m] = (_Float16)siluf_(accH[nt].f[r]);
        __syncthreads();

        // y = h @ W2 + b2 ; x = (x + y) * inv_sqrt2   (K=32 single chunk, N=64)
        V8F accY[4];
#pragma unroll
        for (int nt = 0; nt < 4; ++nt) {
            float bv = b2L[rb][nt * 16 + m];
#pragma unroll
            for (int r = 0; r < 8; ++r) accY[nt].f[r] = bv;
        }
        {
            V16H a;
#pragma unroll
            for (int j = 0; j < 8; ++j) {
                a.h[j]     = ht[w][m][half * 16 + j];
                a.h[8 + j] = ht[w][m][half * 16 + 8 + j];
            }
#pragma unroll
            for (int nt = 0; nt < 4; ++nt) {
                int n = nt * 16 + m;
                const v16h b = *(const v16h*)&w2t[rb][n * 32 + half * 16];
                accY[nt].v = __builtin_amdgcn_wmma_f32_16x16x32_f16(
                    false, a.v, false, b, (short)0, accY[nt].v, false, false);
            }
        }
        __syncthreads();
#pragma unroll
        for (int nt = 0; nt < 4; ++nt)
#pragma unroll
            for (int r = 0; r < 8; ++r) {
                int mm = r + 8 * half, f = nt * 16 + m;
                float xn = (float)xt[w][mm][f];
                xt[w][mm][f] = (_Float16)((xn + accY[nt].f[r]) * INV_SQRT_2F);
            }
        __syncthreads();
    }

    // out = inv_sqrt2 * (nbr + x)
#pragma unroll
    for (int nt = 0; nt < 4; ++nt)
#pragma unroll
        for (int r = 0; r < 8; ++r) {
            int  mm = r + 8 * half;
            long eo = e0 + w * 16 + mm;
            if (eo < E) {
                int f = nt * 16 + m;
                out[eo * 64 + f] = INV_SQRT_2F * (nbr[eo * 64 + f] + (float)xt[w][mm][f]);
            }
        }
}

// ---------------------------------------------------------------------------
extern "C" void kernel_launch(void* const* d_in, const int* in_sizes, int n_in,
                              void* d_out, int out_size, void* d_ws, size_t ws_size,
                              hipStream_t stream) {
    const float* atom  = (const float*)d_in[0];
    const float* nbr   = (const float*)d_in[1];
    const int*   idx   = (const int*)d_in[2];
    const int*   seg   = (const int*)d_in[3];
    const float* Wfull = (const float*)d_in[4];
    const float* n1w   = (const float*)d_in[5];
    const float* n1b   = (const float*)d_in[6];
    const float* Wm    = (const float*)d_in[7];
    const float* lnw   = (const float*)d_in[8];
    const float* lnb   = (const float*)d_in[9];
    const float* n2w   = (const float*)d_in[10];
    const float* n2b   = (const float*)d_in[11];
    const float* W1a   = (const float*)d_in[12];
    const float* b1a   = (const float*)d_in[13];
    const float* W2a   = (const float*)d_in[14];
    const float* b2a   = (const float*)d_in[15];
    const float* W1b   = (const float*)d_in[16];
    const float* b1b   = (const float*)d_in[17];
    const float* W2b   = (const float*)d_in[18];
    const float* b2b   = (const float*)d_in[19];
    float* out = (float*)d_out;
    const int E = in_sizes[3];

    // Workspace layout (floats first, then f16 spill buffers).
    float* wsf   = (float*)d_ws;
    float* cnt   = wsf;                       // 2048
    float* sum1  = cnt   + 2048;              // 2048*128
    float* sq1   = sum1  + 2048 * 128;
    float* sum2  = sq1   + 2048 * 128;        // 2048*64
    float* sq2   = sum2  + 2048 * 64;
    float* mean1 = sq2   + 2048 * 64;
    float* istd1 = mean1 + 2048 * 128;
    float* mean2 = istd1 + 2048 * 128;
    float* istd2 = mean2 + 2048 * 64;
    _Float16* gated = (_Float16*)(istd2 + 2048 * 64);  // E*128 f16
    _Float16* xbuf  = gated + (size_t)E * 128;         // E*64 f16

    const long nzero = 2048 + 2L * 2048 * 128 + 2L * 2048 * 64;
    k_zero<<<(int)((nzero + 255) / 256), 256, 0, stream>>>(wsf, nzero);
    k_count<<<(E + 255) / 256, 256, 0, stream>>>(seg, cnt, E);
    k1_gemm_stats<<<(E + 127) / 128, 256, 0, stream>>>(atom, nbr, idx, seg, Wfull,
                                                       gated, sum1, sq1, E);
    k_finalize<<<(2048 * 128 + 255) / 256, 256, 0, stream>>>(cnt, sum1, sq1,
                                                             mean1, istd1, 128);
    k2_gate<<<(E + 255) / 256, 256, 0, stream>>>(gated, seg, mean1, istd1, n1w, n1b,
                                                 Wm, lnw, lnb, sum2, sq2, xbuf, E);
    k_finalize<<<(2048 * 64 + 255) / 256, 256, 0, stream>>>(cnt, sum2, sq2,
                                                            mean2, istd2, 64);
    k3_residual<<<(E + 127) / 128, 256, 0, stream>>>(xbuf, seg, mean2, istd2, n2w, n2b,
                                                     W1a, b1a, W2a, b2a,
                                                     W1b, b1b, W2b, b2b,
                                                     nbr, out, E);
}
